// TransformerBlock_72275709657180
// MI455X (gfx1250) — matrix operand order repaired and checked by reading, not compiled, not run on hardware
//
#include <hip/hip_runtime.h>
#include <cstdint>

typedef __bf16 bf16_t;
typedef __attribute__((ext_vector_type(16))) __bf16 v16bf;
typedef __attribute__((ext_vector_type(8)))  __bf16 v8bf;
typedef __attribute__((ext_vector_type(8)))  float  v8f;

#define D_MODEL  1024
#define NUM_HEADS 16
#define HEAD_DIM  64
#define D_FF     4096
#define SEQ_LEN  2048
#define BATCH    2
#define NROWS    (BATCH * SEQ_LEN)   // 4096

// ---------------------------------------------------------------------------
// helpers
// ---------------------------------------------------------------------------
__device__ inline v16bf load_frag2(const bf16_t* p0, const bf16_t* p1) {
  v8bf a = *(const v8bf*)p0;
  v8bf b = *(const v8bf*)p1;
  v16bf r;
#pragma unroll
  for (int i = 0; i < 8; ++i) { r[i] = a[i]; r[i + 8] = b[i]; }
  return r;
}

__device__ inline v8f wmma_bf16(v16bf a, v16bf b, v8f c) {
  // D = A(16x32 bf16) * B(32x16 bf16) + C(f32)
  return __builtin_amdgcn_wmma_f32_16x16x32_bf16(false, a, false, b,
                                                 (short)0, c, false, false);
}

// CDNA5 async global->LDS DMA (tracked by ASYNCcnt, no VGPR round-trip).
// LDS operand = wave-relative byte offset (low 32 bits of the flat address).
__device__ inline void async_copy_b128(uint32_t lds_off, const void* gaddr) {
  asm volatile("global_load_async_to_lds_b128 %0, %1, off"
               :: "v"(lds_off), "v"(gaddr)
               : "memory");
}
__device__ inline void wait_async0() {
  asm volatile("s_wait_asynccnt 0x0" ::: "memory");
}
__device__ inline uint32_t lds_off32(const void* p) {
  return (uint32_t)(uintptr_t)p;
}

// ---------------------------------------------------------------------------
// fp32 -> bf16 conversion (weights)
// ---------------------------------------------------------------------------
__global__ void cvt_f32_bf16(const float* __restrict__ src,
                             bf16_t* __restrict__ dst, int n) {
  int i = blockIdx.x * blockDim.x + threadIdx.x;
  int stride = gridDim.x * blockDim.x;
  for (; i < n; i += stride) dst[i] = (bf16_t)src[i];
}

// ---------------------------------------------------------------------------
// RMSNorm: one block per row of D_MODEL f32, writes bf16
// ---------------------------------------------------------------------------
__global__ __launch_bounds__(256) void rmsnorm_kernel(const float* __restrict__ x,
                                                      const float* __restrict__ g,
                                                      bf16_t* __restrict__ out) {
  __shared__ float wsum[8];
  const int row = blockIdx.x;
  const int t = threadIdx.x;
  const float4 xv = *(const float4*)(x + (size_t)row * D_MODEL + t * 4);
  float ss = xv.x * xv.x + xv.y * xv.y + xv.z * xv.z + xv.w * xv.w;
#pragma unroll
  for (int off = 16; off; off >>= 1) ss += __shfl_xor(ss, off, 32);
  if ((t & 31) == 0) wsum[t >> 5] = ss;
  __syncthreads();
  float tot = 0.f;
#pragma unroll
  for (int i = 0; i < 8; ++i) tot += wsum[i];
  const float scale = rsqrtf(tot / (float)D_MODEL + 1e-5f);
  const float4 gv = *(const float4*)(g + t * 4);
  bf16_t ov[4];
  ov[0] = (bf16_t)(xv.x * gv.x * scale);
  ov[1] = (bf16_t)(xv.y * gv.y * scale);
  ov[2] = (bf16_t)(xv.z * gv.z * scale);
  ov[3] = (bf16_t)(xv.w * gv.w * scale);
  *(uint2*)(out + (size_t)row * D_MODEL + t * 4) = *(uint2*)ov;
}

// ---------------------------------------------------------------------------
// GEMM: C[M,N](f32) = A[M,K](bf16) * W[N,K](bf16)^T   (y = x @ W.T)
// 256 threads = 8 waves (4Mx2N). Block tile 128(M) x 128(N), K-step 64.
// Each wave owns a 32x64 output (8 WMMA accumulators, 16 WMMAs / K-step;
// 12 ds_load_b128 feed 8 WMMAs per K=32 chunk via A/B fragment reuse).
// Double-buffered LDS fed by async global->LDS DMA.
// ---------------------------------------------------------------------------
#define GBM 128
#define GBN 128
#define GBK 64
#define GLD 72   // padded LDS row stride (bf16 elems)

__global__ __launch_bounds__(256) void gemm_bf16_kernel(const bf16_t* __restrict__ A,
                                                        const bf16_t* __restrict__ W,
                                                        float* __restrict__ C,
                                                        int M, int N, int K) {
  __shared__ __align__(16) bf16_t As[2][GBM * GLD];
  __shared__ __align__(16) bf16_t Ws[2][GBN * GLD];
  const int t = threadIdx.x;
  const int lane = t & 31, wave = t >> 5;
  const int half = lane >> 4, l15 = lane & 15;
  const int waveM = wave >> 1, waveN = wave & 1;
  const int mblk = blockIdx.y * GBM, nblk = blockIdx.x * GBN;

  v8f acc[2][4];
  {
    v8f z = {};
#pragma unroll
    for (int mi = 0; mi < 2; ++mi)
#pragma unroll
      for (int ni = 0; ni < 4; ++ni) acc[mi][ni] = z;
  }

  auto stage = [&](int buf, int k0) {
    const uint32_t abase = lds_off32(&As[buf][0]);
    const uint32_t wbase = lds_off32(&Ws[buf][0]);
#pragma unroll
    for (int i = 0; i < 4; ++i) {            // 128 rows x 8 chunks
      const int idx = t + i * 256, row = idx >> 3, c = idx & 7;
      async_copy_b128(abase + (uint32_t)(row * GLD + c * 8) * 2,
                      A + (size_t)(mblk + row) * K + k0 + c * 8);
    }
#pragma unroll
    for (int i = 0; i < 4; ++i) {            // 128 rows x 8 chunks
      const int idx = t + i * 256, row = idx >> 3, c = idx & 7;
      async_copy_b128(wbase + (uint32_t)(row * GLD + c * 8) * 2,
                      W + (size_t)(nblk + row) * K + k0 + c * 8);
    }
  };

  const int ar0 = waveM * 32 + l15;   // A rows ar0, ar0+16
  const int br0 = waveN * 64 + l15;   // W rows (output cols) br0 + {0,16,32,48}

  stage(0, 0);
  wait_async0();
  __syncthreads();
  int buf = 0;
  for (int k0 = 0; k0 < K; k0 += GBK) {
    if (k0 + GBK < K) stage(buf ^ 1, k0 + GBK);
    const bf16_t* as = As[buf];
    const bf16_t* ws = Ws[buf];
#pragma unroll
    for (int s = 0; s < 2; ++s) {            // two K=32 chunks
      v16bf af[2], bfm[4];
#pragma unroll
      for (int mi = 0; mi < 2; ++mi) {
        const bf16_t* ap = as + (ar0 + mi * 16) * GLD + s * 32;
        af[mi] = load_frag2(ap + half * 8, ap + half * 8 + 16);
      }
#pragma unroll
      for (int ni = 0; ni < 4; ++ni) {
        const bf16_t* bp = ws + (br0 + ni * 16) * GLD + s * 32;
        bfm[ni] = load_frag2(bp + half * 8, bp + half * 8 + 16);
      }
#pragma unroll
      for (int mi = 0; mi < 2; ++mi)
#pragma unroll
        for (int ni = 0; ni < 4; ++ni)
          acc[mi][ni] = wmma_bf16(af[mi], bfm[ni], acc[mi][ni]);
    }
    wait_async0();
    __syncthreads();
    buf ^= 1;
  }

  const int orow0 = mblk + waveM * 32 + half * 8;
  const int ocol0 = nblk + waveN * 64 + l15;
#pragma unroll
  for (int mi = 0; mi < 2; ++mi)
#pragma unroll
    for (int ni = 0; ni < 4; ++ni)
#pragma unroll
      for (int r = 0; r < 8; ++r)
        C[(size_t)(orow0 + mi * 16 + r) * N + ocol0 + ni * 16] = acc[mi][ni][r];
}

// ---------------------------------------------------------------------------
// RoPE on q,k + split qkv f32 -> q,k,v bf16.  One block per token row.
// ---------------------------------------------------------------------------
__global__ __launch_bounds__(512) void rope_split_kernel(const float* __restrict__ qkv,
                                                         const int* __restrict__ pos,
                                                         bf16_t* __restrict__ Q,
                                                         bf16_t* __restrict__ Kb,
                                                         bf16_t* __restrict__ Vb) {
  const int m = blockIdx.x;
  const int s = m % SEQ_LEN;
  const int t = threadIdx.x;
  const int h = t >> 5, i = t & 31;          // 32 (even,odd) pairs per head
  const int base = h * HEAD_DIM + 2 * i;
  const float* row = qkv + (size_t)m * 3 * D_MODEL;
  const float p = (float)pos[s];
  const float ang = p * __powf(10000.0f, -(float)i / 32.0f);
  float sn, cs;
  __sincosf(ang, &sn, &cs);
  const size_t ob = (size_t)m * D_MODEL;
  const float qe = row[base], qo = row[base + 1];
  Q[ob + base]     = (bf16_t)(qe * cs - qo * sn);
  Q[ob + base + 1] = (bf16_t)(qo * cs + qe * sn);
  const float ke = row[D_MODEL + base], ko = row[D_MODEL + base + 1];
  Kb[ob + base]     = (bf16_t)(ke * cs - ko * sn);
  Kb[ob + base + 1] = (bf16_t)(ko * cs + ke * sn);
  Vb[ob + 2 * t]     = (bf16_t)row[2 * D_MODEL + 2 * t];
  Vb[ob + 2 * t + 1] = (bf16_t)row[2 * D_MODEL + 2 * t + 1];
}

// ---------------------------------------------------------------------------
// Causal flash attention. grid = (S/64, H, B); 128 threads = 4 waves.
// Each wave owns 16 q rows; K/V tiles of 64 kv positions staged in LDS
// (K tile via async DMA; V transposed in software from uint4 loads).
// ---------------------------------------------------------------------------
__global__ __launch_bounds__(128) void attention_kernel(const bf16_t* __restrict__ Q,
                                                        const bf16_t* __restrict__ K,
                                                        const bf16_t* __restrict__ V,
                                                        bf16_t* __restrict__ O) {
  __shared__ __align__(16) bf16_t Ks[64 * 72];      // [kv][dh] row-major
  __shared__ __align__(16) bf16_t Vt[64 * 72];      // [dh][kv] transposed
  __shared__ __align__(16) bf16_t Ps[4][16 * 40];   // per-wave P (16 q x 32 kv)
  const int t = threadIdx.x, lane = t & 31, wave = t >> 5;
  const int half = lane >> 4, l15 = lane & 15;
  const int qblk = blockIdx.x * 64, h = blockIdx.y, b = blockIdx.z;
  const int qw = qblk + wave * 16;
  const size_t rb = (size_t)b * SEQ_LEN;
  const bf16_t* qp = Q + (rb + qw + l15) * D_MODEL + h * HEAD_DIM;
  const v16bf qf0 = load_frag2(qp + half * 8,      qp + half * 8 + 16);
  const v16bf qf1 = load_frag2(qp + 32 + half * 8, qp + 32 + half * 8 + 16);
  float mrow[8], lrow[8];
  v8f acc[4];
#pragma unroll
  for (int r = 0; r < 8; ++r) { mrow[r] = -1e30f; lrow[r] = 0.f; }
#pragma unroll
  for (int d = 0; d < 4; ++d) { v8f z = {}; acc[d] = z; }

  const uint32_t ksbase = lds_off32(&Ks[0]);
  const int ntiles = blockIdx.x + 1;   // causal
  for (int it = 0; it < ntiles; ++it) {
    const int kv0 = it * 64;
    __syncthreads();
    // stage K tile via async DMA: 64 rows x 8 chunks of 16B
#pragma unroll
    for (int i = 0; i < 4; ++i) {
      const int idx = t + i * 128, row = idx >> 3, c = idx & 7;
      async_copy_b128(ksbase + (uint32_t)(row * 72 + c * 8) * 2,
                      K + (rb + kv0 + row) * D_MODEL + h * HEAD_DIM + c * 8);
    }
    // stage V transposed: vector global loads, scalar LDS scatter
#pragma unroll
    for (int i = 0; i < 4; ++i) {
      const int idx = t + i * 128, kv = idx >> 3, c = idx & 7;
      union { uint4 u; bf16_t e[8]; } vv;
      vv.u = *(const uint4*)(V + (rb + kv0 + kv) * D_MODEL + h * HEAD_DIM + c * 8);
#pragma unroll
      for (int j = 0; j < 8; ++j) Vt[(c * 8 + j) * 72 + kv] = vv.e[j];
    }
    wait_async0();
    __syncthreads();

    for (int hk = 0; hk < 2; ++hk) {       // two kv 32-chunks per tile
#pragma unroll
      for (int nt2 = 0; nt2 < 2; ++nt2) {  // two 16-col score tiles per chunk
        const int nt = hk * 2 + nt2;
        const int kcol = nt * 16 + l15;
        v8f sc = {};
        const bf16_t* kp = Ks + kcol * 72;
        v16bf b0 = load_frag2(kp + half * 8,      kp + half * 8 + 16);
        sc = wmma_bf16(qf0, b0, sc);
        v16bf b1 = load_frag2(kp + 32 + half * 8, kp + 32 + half * 8 + 16);
        sc = wmma_bf16(qf1, b1, sc);
        const int colg = kv0 + nt * 16 + l15;
#pragma unroll
        for (int r = 0; r < 8; ++r) {
          const int rowg = qw + r + 8 * half;
          const bool masked = (colg > rowg);
          float v = sc[r] * 0.125f;            // 1/sqrt(64)
          if (masked) v = -1e30f;
          float rmax = v;
#pragma unroll
          for (int off = 1; off < 16; off <<= 1)
            rmax = fmaxf(rmax, __shfl_xor(rmax, off, 32));
          const float newm = fmaxf(mrow[r], rmax);
          const float so = __expf(mrow[r] - newm);
          float p = masked ? 0.f : __expf(v - newm);
          float psum = p;
#pragma unroll
          for (int off = 1; off < 16; off <<= 1)
            psum += __shfl_xor(psum, off, 32);
          lrow[r] = lrow[r] * so + psum;
          mrow[r] = newm;
#pragma unroll
          for (int d = 0; d < 4; ++d) acc[d][r] *= so;
          Ps[wave][(r + 8 * half) * 40 + nt2 * 16 + l15] = (bf16_t)p;
        }
      }
      __syncthreads();
      // P (16x32) * V (32x64) accumulate
      const bf16_t* pw = Ps[wave] + l15 * 40;
      const v16bf af = load_frag2(pw + half * 8, pw + half * 8 + 16);
#pragma unroll
      for (int d = 0; d < 4; ++d) {
        const bf16_t* vp = Vt + (d * 16 + l15) * 72 + hk * 32;
        v16bf bv = load_frag2(vp + half * 8, vp + half * 8 + 16);
        acc[d] = wmma_bf16(af, bv, acc[d]);
      }
      __syncthreads();
    }
  }
#pragma unroll
  for (int d = 0; d < 4; ++d) {
#pragma unroll
    for (int r = 0; r < 8; ++r) {
      const int rowg = qw + r + 8 * half;
      O[(rb + rowg) * D_MODEL + h * HEAD_DIM + d * 16 + l15] =
          (bf16_t)(acc[d][r] / lrow[r]);
    }
  }
}

// ---------------------------------------------------------------------------
// elementwise
// ---------------------------------------------------------------------------
__global__ void swiglu_kernel(const float* __restrict__ h1,
                              const float* __restrict__ h3,
                              bf16_t* __restrict__ g, size_t n) {
  size_t i = blockIdx.x * (size_t)blockDim.x + threadIdx.x;
  const size_t stride = (size_t)gridDim.x * blockDim.x;
  for (; i < n; i += stride) {
    const float a = h1[i];
    const float sig = 1.0f / (1.0f + __expf(-a));
    g[i] = (bf16_t)(a * sig * h3[i]);
  }
}

__global__ void add_kernel(const float* __restrict__ a,
                           const float* __restrict__ b,
                           float* __restrict__ o, size_t n) {
  size_t i = blockIdx.x * (size_t)blockDim.x + threadIdx.x;
  const size_t stride = (size_t)gridDim.x * blockDim.x;
  for (; i < n; i += stride) o[i] = a[i] + b[i];
}

// ---------------------------------------------------------------------------
// host
// ---------------------------------------------------------------------------
extern "C" void kernel_launch(void* const* d_in, const int* in_sizes, int n_in,
                              void* d_out, int out_size, void* d_ws, size_t ws_size,
                              hipStream_t stream) {
  const float* x    = (const float*)d_in[0];
  const int*   pos  = (const int*)d_in[1];
  const float* Wqkv = (const float*)d_in[2];
  const float* Wo   = (const float*)d_in[3];
  const float* w1   = (const float*)d_in[4];
  const float* w2   = (const float*)d_in[5];
  const float* w3   = (const float*)d_in[6];
  const float* g1   = (const float*)d_in[7];
  const float* g2   = (const float*)d_in[8];

  char* base = (char*)d_ws;
  size_t off = 0;
  auto alloc = [&](size_t bytes) {
    void* p = base + off;
    off = (off + bytes + 255) & ~(size_t)255;
    return p;
  };
  const size_t R = NROWS;
  bf16_t* xn    = (bf16_t*)alloc(R * D_MODEL * 2);
  bf16_t* wqkvb = (bf16_t*)alloc((size_t)3 * D_MODEL * D_MODEL * 2);
  bf16_t* wob   = (bf16_t*)alloc((size_t)D_MODEL * D_MODEL * 2);
  bf16_t* w1b   = (bf16_t*)alloc((size_t)D_FF * D_MODEL * 2);
  bf16_t* w2b   = (bf16_t*)alloc((size_t)D_MODEL * D_FF * 2);
  bf16_t* w3b   = (bf16_t*)alloc((size_t)D_FF * D_MODEL * 2);
  float*  qkv   = (float*)alloc(R * 3 * D_MODEL * 4);
  bf16_t* qb    = (bf16_t*)alloc(R * D_MODEL * 2);
  bf16_t* kb    = (bf16_t*)alloc(R * D_MODEL * 2);
  bf16_t* vb    = (bf16_t*)alloc(R * D_MODEL * 2);
  bf16_t* ab    = (bf16_t*)alloc(R * D_MODEL * 2);
  float*  proj  = (float*)alloc(R * D_MODEL * 4);
  float*  hbuf  = (float*)alloc(R * D_MODEL * 4);
  bf16_t* hn    = (bf16_t*)alloc(R * D_MODEL * 2);
  float*  h1    = (float*)alloc(R * (size_t)D_FF * 4);
  float*  h3    = (float*)alloc(R * (size_t)D_FF * 4);
  bf16_t* gateb = (bf16_t*)alloc(R * (size_t)D_FF * 2);
  float*  ffn   = (float*)alloc(R * D_MODEL * 4);

  cvt_f32_bf16<<<1024, 256, 0, stream>>>(Wqkv, wqkvb, 3 * D_MODEL * D_MODEL);
  cvt_f32_bf16<<<1024, 256, 0, stream>>>(Wo,   wob,   D_MODEL * D_MODEL);
  cvt_f32_bf16<<<1024, 256, 0, stream>>>(w1,   w1b,   D_FF * D_MODEL);
  cvt_f32_bf16<<<1024, 256, 0, stream>>>(w2,   w2b,   D_MODEL * D_FF);
  cvt_f32_bf16<<<1024, 256, 0, stream>>>(w3,   w3b,   D_FF * D_MODEL);

  rmsnorm_kernel<<<R, 256, 0, stream>>>(x, g1, xn);
  gemm_bf16_kernel<<<dim3(3 * D_MODEL / GBN, R / GBM), 256, 0, stream>>>(
      xn, wqkvb, qkv, (int)R, 3 * D_MODEL, D_MODEL);
  rope_split_kernel<<<R, 512, 0, stream>>>(qkv, pos, qb, kb, vb);
  attention_kernel<<<dim3(SEQ_LEN / 64, NUM_HEADS, BATCH), 128, 0, stream>>>(
      qb, kb, vb, ab);
  gemm_bf16_kernel<<<dim3(D_MODEL / GBN, R / GBM), 256, 0, stream>>>(
      ab, wob, proj, (int)R, D_MODEL, D_MODEL);
  add_kernel<<<2048, 256, 0, stream>>>(x, proj, hbuf, R * (size_t)D_MODEL);
  rmsnorm_kernel<<<R, 256, 0, stream>>>(hbuf, g2, hn);
  gemm_bf16_kernel<<<dim3(D_FF / GBN, R / GBM), 256, 0, stream>>>(
      hn, w1b, h1, (int)R, D_FF, D_MODEL);
  gemm_bf16_kernel<<<dim3(D_FF / GBN, R / GBM), 256, 0, stream>>>(
      hn, w3b, h3, (int)R, D_FF, D_MODEL);
  swiglu_kernel<<<4096, 256, 0, stream>>>(h1, h3, gateb, R * (size_t)D_FF);
  gemm_bf16_kernel<<<dim3(D_MODEL / GBN, R / GBM), 256, 0, stream>>>(
      gateb, w2b, ffn, (int)R, D_MODEL, D_FF);
  add_kernel<<<2048, 256, 0, stream>>>(hbuf, ffn, (float*)d_out, R * (size_t)D_MODEL);
}
